// MoESwiGLU_72962904424749
// MI455X (gfx1250) — matrix operand order repaired and checked by reading, not compiled, not run on hardware
//
#include <hip/hip_runtime.h>
#include <hip/hip_bf16.h>

// Problem constants (from reference): B=4, T=2048, C=512, H=1024, E=8, K=2
#define BT    8192
#define C_DIM 512
#define H_DIM 1024
#define E_NUM 8
#define TM    64          // tokens per expert tile (4 row-tiles of 16)

typedef __attribute__((ext_vector_type(16))) __bf16     v16bf;
typedef __attribute__((ext_vector_type(4)))  __bf16     v4bf;
typedef __attribute__((ext_vector_type(8)))  float      v8f;
typedef __attribute__((ext_vector_type(4)))  unsigned   u32x4;

union ABFrag {
    v16bf  v;
    u32x4  q[2];
    __bf16 h[16];
};

// ---------------------------------------------------------------------------
// Kernel 0: zero output + workspace header
// ---------------------------------------------------------------------------
__global__ __launch_bounds__(256) void moe_zero_kernel(float* __restrict__ out,
                                                       long n, int* __restrict__ wshdr) {
    long i = (long)blockIdx.x * 256 + threadIdx.x;
    if (i < n) out[i] = 0.0f;
    if (blockIdx.x == 0 && threadIdx.x < 32) wshdr[threadIdx.x] = 0;
}

// ---------------------------------------------------------------------------
// Kernel 0b: fp32 [R x Cc] -> transposed bf16 [Cc x R], per expert (blockIdx.z)
// Tiled via LDS so both global sides stay coalesced.
// ---------------------------------------------------------------------------
__global__ __launch_bounds__(256) void moe_transpose_cvt_kernel(
    const float* __restrict__ in, __bf16* __restrict__ outp, int R, int Cc)
{
    __shared__ float t[32][33];
    const long base = (long)blockIdx.z * R * Cc;
    const int r0 = blockIdx.y * 32;
    const int c0 = blockIdx.x * 32;
    const int tx = threadIdx.x, ty = threadIdx.y;
#pragma unroll
    for (int i = 0; i < 4; ++i)
        t[ty + 8 * i][tx] = in[base + (long)(r0 + ty + 8 * i) * Cc + c0 + tx];
    __syncthreads();
#pragma unroll
    for (int i = 0; i < 4; ++i)
        outp[base + (long)(c0 + ty + 8 * i) * R + r0 + tx] = (__bf16)t[tx][ty + 8 * i];
}

// ---------------------------------------------------------------------------
// Kernel 1: router — logits, top-2 softmax weights, expert binning, aux stats
// One wave32 per token; 8 waves (256 threads) per block.
// ---------------------------------------------------------------------------
__global__ __launch_bounds__(256) void moe_router_kernel(
    const float* __restrict__ x, const float* __restrict__ wr,
    int* __restrict__ cnt, float* __restrict__ prob_sum, float* __restrict__ top1cnt,
    int* __restrict__ bins, float* __restrict__ binw)
{
    __shared__ float s_prob[E_NUM];
    __shared__ float s_top1[E_NUM];
    const int tid = threadIdx.x;
    if (tid < E_NUM) { s_prob[tid] = 0.0f; s_top1[tid] = 0.0f; }
    __syncthreads();

    const int lane = tid & 31;
    const int wave = tid >> 5;
    const int t = blockIdx.x * 8 + wave;

    float acc[E_NUM];
#pragma unroll
    for (int e = 0; e < E_NUM; ++e) acc[e] = 0.0f;

    for (int c = lane; c < C_DIM; c += 32) {
        float xv = x[(long)t * C_DIM + c];
#pragma unroll
        for (int e = 0; e < E_NUM; ++e)
            acc[e] += xv * wr[c * E_NUM + e];
    }
#pragma unroll
    for (int e = 0; e < E_NUM; ++e) {
        for (int off = 16; off > 0; off >>= 1)
            acc[e] += __shfl_xor(acc[e], off, 32);
    }

    if (lane == 0) {
        // top-2 (first max on ties, matching jax top_k)
        int i0 = 0;
#pragma unroll
        for (int e = 1; e < E_NUM; ++e) if (acc[e] > acc[i0]) i0 = e;
        int i1 = (i0 == 0) ? 1 : 0;
#pragma unroll
        for (int e = 0; e < E_NUM; ++e) if (e != i0 && acc[e] > acc[i1]) i1 = e;
        float v0 = acc[i0], v1 = acc[i1];
        float e1 = __expf(v1 - v0);
        float inv01 = __builtin_amdgcn_rcpf(1.0f + e1);
        float w0 = inv01;
        float w1 = e1 * inv01;

        int p0 = atomicAdd(&cnt[i0], 1);
        bins[i0 * BT + p0] = t;  binw[i0 * BT + p0] = w0;
        int p1 = atomicAdd(&cnt[i1], 1);
        bins[i1 * BT + p1] = t;  binw[i1 * BT + p1] = w1;

        // full softmax over E for aux loss
        float p[E_NUM], s = 0.0f;
#pragma unroll
        for (int e = 0; e < E_NUM; ++e) { p[e] = __expf(acc[e] - v0); s += p[e]; }
        float inv = __builtin_amdgcn_rcpf(s);
#pragma unroll
        for (int e = 0; e < E_NUM; ++e) atomicAdd(&s_prob[e], p[e] * inv);
        atomicAdd(&s_top1[i0], 1.0f);
    }
    __syncthreads();
    if (tid < E_NUM) {
        atomicAdd(&prob_sum[tid], s_prob[tid]);
        atomicAdd(&top1cnt[tid], s_top1[tid]);
    }
}

// ---------------------------------------------------------------------------
// Kernel 2: expert SwiGLU via WMMA bf16 (f32 accumulate)
// grid = E_NUM * (BT/TM) blocks; 256 threads = 8 waves; 192 KB LDS.
// K-loop outer, col-tiles inner: A fragments loaded once per K-step and
// reused by 4 col-tiles; 128 persistent acc VGPRs; no spills.
// ---------------------------------------------------------------------------
__device__ __forceinline__ v8f wmma_bf16(const v16bf& a, const v16bf& b, v8f c) {
    return __builtin_amdgcn_wmma_f32_16x16x32_bf16(false, a, false, b, (short)0, c,
                                                   false, false);
}

__global__ __launch_bounds__(256, 1) void moe_expert_kernel(
    const float*  __restrict__ x,
    const __bf16* __restrict__ wgT,     // [E][H][C] bf16 (transposed gate)
    const __bf16* __restrict__ wuT,     // [E][H][C] bf16 (transposed up)
    const __bf16* __restrict__ wdT,     // [E][C][H] bf16 (transposed down)
    const int*    __restrict__ cnt,
    const int*    __restrict__ bins,
    const float*  __restrict__ binw,
    float*        __restrict__ out)
{
    __shared__ __bf16 xA[TM][C_DIM];      //  64 KB: token tile activations
    __shared__ __bf16 act[TM][H_DIM];     // 128 KB: u, then silu(g)*u

    const int tilesPerExpert = BT / TM;   // 128
    const int e     = blockIdx.x / tilesPerExpert;
    const int tile  = blockIdx.x % tilesPerExpert;
    const int start = tile * TM;
    const int n_e   = cnt[e];
    if (start >= n_e) return;             // uniform per block
    int rows = n_e - start; if (rows > TM) rows = TM;

    const int tid  = threadIdx.x;
    const int lane = tid & 31;
    const int wave = tid >> 5;
    const int hi   = lane >> 4;           // lane half (0/1)
    const int nl   = lane & 15;

    // ---- gather token rows of x into LDS as bf16 (float4 vectorized) ----
    for (int i = tid; i < TM * (C_DIM / 4); i += 256) {
        int r  = i >> 7;                  // / (C_DIM/4)
        int c4 = (i & (C_DIM / 4 - 1)) * 4;
        v4bf tbf = {};
        if (r < rows) {
            int tok = bins[e * BT + start + r];
            float4 v = *(const float4*)(x + (long)tok * C_DIM + c4);
            tbf[0] = (__bf16)v.x; tbf[1] = (__bf16)v.y;
            tbf[2] = (__bf16)v.z; tbf[3] = (__bf16)v.w;
        }
        *(v4bf*)&xA[r][c4] = tbf;
    }

    const __bf16* wgE = wgT + (long)e * H_DIM * C_DIM;
    const __bf16* wuE = wuT + (long)e * H_DIM * C_DIM;
    const __bf16* wdE = wdT + (long)e * C_DIM * H_DIM;

    __syncthreads();

    // ---- stage 1, pass U: u = x@Wu -> act (bf16) ----
    for (int jg = 0; jg < 2; ++jg) {                // 2 groups of 4 col-tiles
        const int h0g = (wave * 8 + jg * 4) * 16;
        v8f u[4][4];
#pragma unroll
        for (int j = 0; j < 4; ++j)
#pragma unroll
            for (int rt = 0; rt < 4; ++rt) u[j][rt] = {};

#pragma unroll 1
        for (int kt = 0; kt < C_DIM / 32; ++kt) {
            const int kb = kt * 32;
            ABFrag a[4];
#pragma unroll
            for (int rt = 0; rt < 4; ++rt) {
                a[rt].q[0] = *(const u32x4*)&xA[nl + rt * 16][kb + hi * 8];
                a[rt].q[1] = *(const u32x4*)&xA[nl + rt * 16][kb + 16 + hi * 8];
            }
#pragma unroll
            for (int j = 0; j < 4; ++j) {
                const __bf16* p = wuE + (long)(h0g + j * 16 + nl) * C_DIM + hi * 8 + kb;
                ABFrag b;
                b.q[0] = *(const u32x4*)p;
                b.q[1] = *(const u32x4*)(p + 16);
#pragma unroll
                for (int rt = 0; rt < 4; ++rt)
                    u[j][rt] = wmma_bf16(a[rt].v, b.v, u[j][rt]);
            }
        }
#pragma unroll
        for (int j = 0; j < 4; ++j) {
            const int colBase = h0g + j * 16 + nl;
#pragma unroll
            for (int rt = 0; rt < 4; ++rt)
#pragma unroll
                for (int r = 0; r < 8; ++r)
                    act[rt * 16 + hi * 8 + r][colBase] = (__bf16)u[j][rt][r];
        }
    }

    // ---- stage 1, pass G: g = x@Wg ; act = silu(g) * act ----
    // Same lane re-reads the addresses it wrote (same-wave DS ordering, no barrier).
    for (int jg = 0; jg < 2; ++jg) {
        const int h0g = (wave * 8 + jg * 4) * 16;
        v8f g[4][4];
#pragma unroll
        for (int j = 0; j < 4; ++j)
#pragma unroll
            for (int rt = 0; rt < 4; ++rt) g[j][rt] = {};

#pragma unroll 1
        for (int kt = 0; kt < C_DIM / 32; ++kt) {
            const int kb = kt * 32;
            ABFrag a[4];
#pragma unroll
            for (int rt = 0; rt < 4; ++rt) {
                a[rt].q[0] = *(const u32x4*)&xA[nl + rt * 16][kb + hi * 8];
                a[rt].q[1] = *(const u32x4*)&xA[nl + rt * 16][kb + 16 + hi * 8];
            }
#pragma unroll
            for (int j = 0; j < 4; ++j) {
                const __bf16* p = wgE + (long)(h0g + j * 16 + nl) * C_DIM + hi * 8 + kb;
                ABFrag b;
                b.q[0] = *(const u32x4*)p;
                b.q[1] = *(const u32x4*)(p + 16);
#pragma unroll
                for (int rt = 0; rt < 4; ++rt)
                    g[j][rt] = wmma_bf16(a[rt].v, b.v, g[j][rt]);
            }
        }
#pragma unroll
        for (int j = 0; j < 4; ++j) {
            const int colBase = h0g + j * 16 + nl;
#pragma unroll
            for (int rt = 0; rt < 4; ++rt) {
#pragma unroll
                for (int r = 0; r < 8; ++r) {
                    int m = rt * 16 + hi * 8 + r;
                    float uv = (float)act[m][colBase];
                    float gv = g[j][rt][r];
                    float sv = gv * __builtin_amdgcn_rcpf(1.0f + __expf(-gv));
                    act[m][colBase] = (__bf16)(sv * uv);
                }
            }
        }
    }
    __syncthreads();

    // ---- stage 2: down-projection over K = H_DIM, scale, atomic-add ----
    {
        v8f d[4][4];                                // [C col-tile][row-tile]
#pragma unroll
        for (int j = 0; j < 4; ++j)
#pragma unroll
            for (int rt = 0; rt < 4; ++rt) d[j][rt] = {};

#pragma unroll 1
        for (int kt = 0; kt < H_DIM / 32; ++kt) {
            const int kb = kt * 32;
            ABFrag a[4];
#pragma unroll
            for (int rt = 0; rt < 4; ++rt) {
                a[rt].q[0] = *(const u32x4*)&act[nl + rt * 16][kb + hi * 8];
                a[rt].q[1] = *(const u32x4*)&act[nl + rt * 16][kb + 16 + hi * 8];
            }
#pragma unroll
            for (int j = 0; j < 4; ++j) {
                const int c0 = (wave * 4 + j) * 16;
                const __bf16* p = wdE + (long)(c0 + nl) * H_DIM + hi * 8 + kb;
                ABFrag b;
                b.q[0] = *(const u32x4*)p;
                b.q[1] = *(const u32x4*)(p + 16);
#pragma unroll
                for (int rt = 0; rt < 4; ++rt)
                    d[j][rt] = wmma_bf16(a[rt].v, b.v, d[j][rt]);
            }
        }
#pragma unroll
        for (int rt = 0; rt < 4; ++rt) {
#pragma unroll
            for (int r = 0; r < 8; ++r) {
                int m = rt * 16 + hi * 8 + r;
                if (m < rows) {
                    int   tok = bins[e * BT + start + m];
                    float wgt = binw[e * BT + start + m];
#pragma unroll
                    for (int j = 0; j < 4; ++j) {
                        int c0 = (wave * 4 + j) * 16;
                        atomicAdd(&out[(long)tok * C_DIM + c0 + nl], d[j][rt][r] * wgt);
                    }
                }
            }
        }
    }
}

// ---------------------------------------------------------------------------
// Kernel 3: aux loss scalar
// ---------------------------------------------------------------------------
__global__ void moe_aux_kernel(const float* __restrict__ prob_sum,
                               const float* __restrict__ top1cnt,
                               float* __restrict__ out_aux) {
    if (threadIdx.x == 0 && blockIdx.x == 0) {
        float s = 0.0f;
        for (int e = 0; e < E_NUM; ++e)
            s += (prob_sum[e] / (float)BT) * (top1cnt[e] / (float)BT);
        out_aux[0] = (float)E_NUM * s;
    }
}

// ---------------------------------------------------------------------------
extern "C" void kernel_launch(void* const* d_in, const int* in_sizes, int n_in,
                              void* d_out, int out_size, void* d_ws, size_t ws_size,
                              hipStream_t stream) {
    const float* x        = (const float*)d_in[0];   // [B,T,C]
    const float* w_router = (const float*)d_in[1];   // [C,E]
    const float* w_gate   = (const float*)d_in[2];   // [E,C,H]
    const float* w_up     = (const float*)d_in[3];   // [E,C,H]
    const float* w_down   = (const float*)d_in[4];   // [E,H,C]
    float* out = (float*)d_out;                      // [B,T,C] + 1 (aux)

    // workspace layout
    int*   cnt      = (int*)d_ws;                      // [8]
    float* prob_sum = (float*)d_ws + 8;                // [8]
    float* top1cnt  = (float*)d_ws + 16;               // [8]
    int*   bins     = (int*)d_ws + 32;                 // [E_NUM * BT]
    float* binw     = (float*)d_ws + 32 + E_NUM * BT;  // [E_NUM * BT]
    const long wElems = (long)E_NUM * C_DIM * H_DIM;   // 4,194,304 per tensor
    __bf16* wgT = (__bf16*)((float*)d_ws + 32 + 2 * E_NUM * BT);
    __bf16* wuT = wgT + wElems;
    __bf16* wdT = wuT + wElems;                        // total ws ~24.5 MB

    const long n_out = (long)BT * C_DIM;               // 4,194,304

    moe_zero_kernel<<<(int)((n_out + 255) / 256), 256, 0, stream>>>(out, n_out, (int*)d_ws);

    // pre-transpose + convert weights to bf16 (N-major for WMMA B fragments)
    dim3 tb(32, 8);
    moe_transpose_cvt_kernel<<<dim3(H_DIM / 32, C_DIM / 32, E_NUM), tb, 0, stream>>>(
        w_gate, wgT, C_DIM, H_DIM);
    moe_transpose_cvt_kernel<<<dim3(H_DIM / 32, C_DIM / 32, E_NUM), tb, 0, stream>>>(
        w_up, wuT, C_DIM, H_DIM);
    moe_transpose_cvt_kernel<<<dim3(C_DIM / 32, H_DIM / 32, E_NUM), tb, 0, stream>>>(
        w_down, wdT, H_DIM, C_DIM);

    moe_router_kernel<<<BT / 8, 256, 0, stream>>>(x, w_router, cnt, prob_sum, top1cnt,
                                                  bins, binw);

    moe_expert_kernel<<<E_NUM * (BT / TM), 256, 0, stream>>>(x, wgT, wuT, wdT,
                                                             cnt, bins, binw, out);

    moe_aux_kernel<<<1, 32, 0, stream>>>(prob_sum, top1cnt, out + n_out);
}
